// Attention_52639119180555
// MI455X (gfx1250) — matrix operand order repaired and checked by reading, not compiled, not run on hardware
//
#include <hip/hip_runtime.h>

// ---------------------------------------------------------------------------
// MI455X / gfx1250 implementation of the masked-attention reference.
// All matmuls (projections, QK^T, attn*V) run on V_WMMA_F32_16X16X32_F16
// (f16 operands, f32 accumulate). wave32 layouts per CDNA5 ISA 7.12.2.
// Projection GEMM register-blocks 32x64 per wave: 8 WMMA per K-step with
// 6 fragment loads (A reused 4x, B reused 2x); L2 (192MB) holds W + acts.
// ---------------------------------------------------------------------------

typedef __attribute__((ext_vector_type(16))) _Float16 v16h;
typedef __attribute__((ext_vector_type(8)))  float    v8f;

#define B_      64
#define SEQ     482
#define DIM     512
#define QKVC    1536
#define HEADS   8
#define DHEAD   64
#define ATT_SCALE 0.125f   // 64^-0.5
#define NEG_BIG (-3.0e38f)

__device__ __forceinline__ v8f wmma_f16(v16h a, v16h b, v8f c) {
    return __builtin_amdgcn_wmma_f32_16x16x32_f16(
        /*neg_a=*/false, a, /*neg_b=*/false, b,
        /*c_mod=*/(short)0, c, /*reuse_a=*/false, /*reuse_b=*/false);
}

// ---------------------------------------------------------------------------
// Generic GEMM: C[M,N] = A[M,K] * B[N,K]^T (+ bias), optional |B|.
// One wave per 32x64 output tile (2 M-tiles x 4 N-tiles, 8 accumulators).
// Requires: M % 32 == 0, N % 64 == 0, K % 32 == 0.
// A-fragment (16-bit 16x32): lane holds row M=lane&15; element e=(v,s):
//   K = (lane>=16?16:0) + 2v + s
// B-fragment (16-bit 32x16): lane holds col N=lane&15; element e=(v,s):
//   K = (lane>=16?16:0) + 2v + s
// C/D f32 16x16: vgpr g -> row g + 8*(lane>=16), col = lane&15.
// ---------------------------------------------------------------------------
__global__ __launch_bounds__(256) void gemm_wmma(
    const float* __restrict__ A, const float* __restrict__ Bw,
    const float* __restrict__ bias, float* __restrict__ C,
    int M, int N, int K, int absB)
{
    int wave = threadIdx.x >> 5;
    int lane = threadIdx.x & 31;
    int hf   = lane >> 4;
    int l15  = lane & 15;
    int nt4  = N >> 6;                         // groups of four 16-col tiles
    long w     = (long)blockIdx.x * 8 + wave;
    long total = (long)(M >> 5) * nt4;
    if (w >= total) return;
    int tm = (int)(w / nt4) << 5;              // 32-row block
    int tn = (int)(w % nt4) << 6;              // 64-col block

    const float* ar0 = A  + (size_t)(tm + l15) * K;
    const float* ar1 = ar0 + (size_t)16 * K;
    const float* br0 = Bw + (size_t)(tn + l15) * K;
    const float* br1 = br0 + (size_t)16 * K;
    const float* br2 = br0 + (size_t)32 * K;
    const float* br3 = br0 + (size_t)48 * K;

    v8f acc[8] = {};                           // [mi*4 + t]
    for (int k0 = 0; k0 < K; k0 += 32) {
        if (k0 + 128 < K) {                    // stream-prefetch A rows
            __builtin_prefetch(ar0 + k0 + 128, 0, 0);
            __builtin_prefetch(ar1 + k0 + 128, 0, 0);
        }
        v16h a0, a1, b0, b1, b2, b3;
#pragma unroll
        for (int v = 0; v < 8; ++v) {
            int ka = k0 + hf * 16 + (v << 1);
            float2 fa0 = *(const float2*)(ar0 + ka);
            float2 fa1 = *(const float2*)(ar1 + ka);
            a0[2 * v] = (_Float16)fa0.x; a0[2 * v + 1] = (_Float16)fa0.y;
            a1[2 * v] = (_Float16)fa1.x; a1[2 * v + 1] = (_Float16)fa1.y;

            int kb = k0 + hf * 16 + (v << 1);
            float2 g0 = *(const float2*)(br0 + kb);
            float2 g1 = *(const float2*)(br1 + kb);
            float2 g2 = *(const float2*)(br2 + kb);
            float2 g3 = *(const float2*)(br3 + kb);
            if (absB) {
                g0.x = fabsf(g0.x); g0.y = fabsf(g0.y);
                g1.x = fabsf(g1.x); g1.y = fabsf(g1.y);
                g2.x = fabsf(g2.x); g2.y = fabsf(g2.y);
                g3.x = fabsf(g3.x); g3.y = fabsf(g3.y);
            }
            b0[2 * v] = (_Float16)g0.x; b0[2 * v + 1] = (_Float16)g0.y;
            b1[2 * v] = (_Float16)g1.x; b1[2 * v + 1] = (_Float16)g1.y;
            b2[2 * v] = (_Float16)g2.x; b2[2 * v + 1] = (_Float16)g2.y;
            b3[2 * v] = (_Float16)g3.x; b3[2 * v + 1] = (_Float16)g3.y;
        }
        acc[0] = wmma_f16(a0, b0, acc[0]);
        acc[1] = wmma_f16(a0, b1, acc[1]);
        acc[2] = wmma_f16(a0, b2, acc[2]);
        acc[3] = wmma_f16(a0, b3, acc[3]);
        acc[4] = wmma_f16(a1, b0, acc[4]);
        acc[5] = wmma_f16(a1, b1, acc[5]);
        acc[6] = wmma_f16(a1, b2, acc[6]);
        acc[7] = wmma_f16(a1, b3, acc[7]);
    }
#pragma unroll
    for (int t = 0; t < 4; ++t) {
        int col = tn + t * 16 + l15;
        float bv = bias ? bias[col] : 0.0f;
#pragma unroll
        for (int mi = 0; mi < 2; ++mi) {
#pragma unroll
            for (int g = 0; g < 8; ++g) {
                int row = tm + mi * 16 + g + hf * 8;
                C[(size_t)row * N + col] = acc[mi * 4 + t][g] + bv;
            }
        }
    }
}

// ---------------------------------------------------------------------------
// inv[b][c] = 1 / (1e-6 + max_n qkv_m[b][n][c])
// ---------------------------------------------------------------------------
__global__ __launch_bounds__(256) void colinv_kernel(
    const float* __restrict__ qkvm, float* __restrict__ inv)
{
    int idx = blockIdx.x * 256 + threadIdx.x;
    if (idx >= B_ * QKVC) return;
    int b = idx / QKVC, c = idx % QKVC;
    const float* p = qkvm + (size_t)b * SEQ * QKVC + c;
    float mx = NEG_BIG;
    for (int n = 0; n < SEQ; ++n) mx = fmaxf(mx, p[(size_t)n * QKVC]);
    inv[idx] = 1.0f / (1e-6f + mx);
}

// ---------------------------------------------------------------------------
// updated[b*SEQ+n] = (n==0) ? 1 : (mean_c mask[b][n][c] > 0)
// ---------------------------------------------------------------------------
__global__ __launch_bounds__(256) void updated_kernel(
    const float* __restrict__ mask, float* __restrict__ upd)
{
    int t = blockIdx.x;                 // b*SEQ + n
    int n = t % SEQ;
    __shared__ float red[8];
    const float* p = mask + (size_t)t * DIM;
    float s = 0.0f;
    for (int c = threadIdx.x; c < DIM; c += 256) s += p[c];
#pragma unroll
    for (int m = 1; m < 32; m <<= 1) s += __shfl_xor(s, m, 32);
    if ((threadIdx.x & 31) == 0) red[threadIdx.x >> 5] = s;
    __syncthreads();
    if (threadIdx.x == 0) {
        float tot = 0.0f;
        for (int i = 0; i < 8; ++i) tot += red[i];
        upd[t] = (n == 0) ? 1.0f : (tot > 0.0f ? 1.0f : 0.0f);
    }
}

// ---------------------------------------------------------------------------
// B-fragment of k_hat^T for S = q_hat * k_hat^T (k_hat row-major [key, k]).
// ---------------------------------------------------------------------------
__device__ __forceinline__ v16h bfrag_k(
    const float* __restrict__ qk, const float* __restrict__ qkm,
    const float* __restrict__ binv, int ck, int key, int kbase, int hf)
{
    v16h r;
    if (key < SEQ) {
        const float* p  = qk  + (size_t)key * QKVC + ck;
        const float* pm = qkm + (size_t)key * QKVC + ck;
#pragma unroll
        for (int v = 0; v < 8; ++v) {
            int k = kbase + hf * 16 + 2 * v;
            r[2 * v]     = (_Float16)(p[k]     * pm[k]     * binv[ck + k]);
            r[2 * v + 1] = (_Float16)(p[k + 1] * pm[k + 1] * binv[ck + k + 1]);
        }
    } else {
#pragma unroll
        for (int e = 0; e < 16; ++e) r[e] = (_Float16)0.0f;
    }
    return r;
}

// B-fragments of V-side (row-major [key, d]): bv = v*v_m, bm = v_m.
__device__ __forceinline__ void bfrag_v(
    const float* __restrict__ qk, const float* __restrict__ qkm,
    float inv_d, int cv, int jb, int t, int hf, int l15,
    v16h& bv, v16h& bm)
{
    int d = t * 16 + l15;
#pragma unroll
    for (int v = 0; v < 8; ++v) {
#pragma unroll
        for (int s = 0; s < 2; ++s) {
            int key = jb + hf * 16 + 2 * v + s;
            float vm = 0.0f, vv = 0.0f;
            if (key < SEQ) {
                size_t idx = (size_t)key * QKVC + cv + d;
                vm = qkm[idx] * inv_d;
                vv = qk[idx];
            }
            bv[2 * v + s] = (_Float16)(vm * vv);
            bm[2 * v + s] = (_Float16)vm;
        }
    }
}

// ---------------------------------------------------------------------------
// Flash-style attention. One wave = 16 query rows of one (b,h).
// out = softmax(q_hat k_hat^T * scale) @ (v_m*v);  m = attn @ v_m.
// Both scaled by updated[row] on store.
// ---------------------------------------------------------------------------
__global__ __launch_bounds__(256) void attn_kernel(
    const float* __restrict__ qkv, const float* __restrict__ qkvm,
    const float* __restrict__ inv, const float* __restrict__ upd,
    float* __restrict__ out_attn, float* __restrict__ m_attn)
{
    __shared__ _Float16 plds[8][16][32];
    int wave = threadIdx.x >> 5;
    int lane = threadIdx.x & 31;
    int hf   = lane >> 4;
    int l15  = lane & 15;
    int bh = blockIdx.x >> 2;
    int b  = bh >> 3;
    int h  = bh & 7;
    int mb = ((blockIdx.x & 3) << 3) + wave;     // 0..31, need 0..30
    if (mb >= 31) return;
    int m0 = mb << 4;

    const float* qk   = qkv  + (size_t)b * SEQ * QKVC;
    const float* qkm  = qkvm + (size_t)b * SEQ * QKVC;
    const float* binv = inv  + (size_t)b * QKVC;
    int cq = h * DHEAD;
    int ck = DIM + h * DHEAD;
    int cv = 2 * DIM + h * DHEAD;

    // q_hat A-fragments for K=0..31 and 32..63 (persistent)
    int qrow = m0 + l15; if (qrow > SEQ - 1) qrow = SEQ - 1;
    const float* qrp  = qk  + (size_t)qrow * QKVC + cq;
    const float* qrpm = qkm + (size_t)qrow * QKVC + cq;
    v16h aq0, aq1;
#pragma unroll
    for (int v = 0; v < 8; ++v) {
#pragma unroll
        for (int s = 0; s < 2; ++s) {
            int k = hf * 16 + (v << 1) + s;
            aq0[2 * v + s] = (_Float16)(qrp[k]      * qrpm[k]      * binv[cq + k]);
            aq1[2 * v + s] = (_Float16)(qrp[32 + k] * qrpm[32 + k] * binv[cq + 32 + k]);
        }
    }

    v8f ao0 = {}, ao1 = {}, ao2 = {}, ao3 = {};
    v8f am0 = {}, am1 = {}, am2 = {}, am3 = {};
    float rm[8], rs[8];
#pragma unroll
    for (int g = 0; g < 8; ++g) { rm[g] = NEG_BIG; rs[g] = 0.0f; }

    for (int jb = 0; jb < SEQ; jb += 32) {
        // S tiles: keys [jb, jb+16) and [jb+16, jb+32)
        v8f s0 = {}, s1 = {};
        s0 = wmma_f16(aq0, bfrag_k(qk, qkm, binv, ck, jb + l15, 0, hf), s0);
        s0 = wmma_f16(aq1, bfrag_k(qk, qkm, binv, ck, jb + l15, 32, hf), s0);
        s1 = wmma_f16(aq0, bfrag_k(qk, qkm, binv, ck, jb + 16 + l15, 0, hf), s1);
        s1 = wmma_f16(aq1, bfrag_k(qk, qkm, binv, ck, jb + 16 + l15, 32, hf), s1);

        bool val0 = (jb + l15) < SEQ;
        bool val1 = (jb + 16 + l15) < SEQ;
        float p0[8], p1[8];
#pragma unroll
        for (int g = 0; g < 8; ++g) {
            float x0 = val0 ? s0[g] * ATT_SCALE : NEG_BIG;
            float x1 = val1 ? s1[g] * ATT_SCALE : NEG_BIG;
            float mx = fmaxf(x0, x1);
#pragma unroll
            for (int msk = 1; msk < 16; msk <<= 1) mx = fmaxf(mx, __shfl_xor(mx, msk, 32));
            float nm = fmaxf(rm[g], mx);
            float f  = __expf(rm[g] - nm);
            rm[g] = nm;
            p0[g] = __expf(x0 - nm);
            p1[g] = __expf(x1 - nm);
            float ps = p0[g] + p1[g];
#pragma unroll
            for (int msk = 1; msk < 16; msk <<= 1) ps += __shfl_xor(ps, msk, 32);
            rs[g] = rs[g] * f + ps;
            ao0[g] *= f; ao1[g] *= f; ao2[g] *= f; ao3[g] *= f;
            am0[g] *= f; am1[g] *= f; am2[g] *= f; am3[g] *= f;
        }

        // P (C-layout) -> LDS -> A-fragment layout
#pragma unroll
        for (int g = 0; g < 8; ++g) {
            plds[wave][g + hf * 8][l15]      = (_Float16)p0[g];
            plds[wave][g + hf * 8][16 + l15] = (_Float16)p1[g];
        }
        asm volatile("s_wait_dscnt 0" ::: "memory");
        v16h pa;
#pragma unroll
        for (int v = 0; v < 8; ++v) {
#pragma unroll
            for (int s = 0; s < 2; ++s) {
                int k = hf * 16 + (v << 1) + s;
                pa[2 * v + s] = plds[wave][l15][k];
            }
        }

        // out += P @ (v*v_m);  m += P @ v_m   (4 d-tiles of 16)
        v16h bv, bm;
        float ivd = binv[cv + 0 * 16 + l15];
        bfrag_v(qk, qkm, ivd, cv, jb, 0, hf, l15, bv, bm);
        ao0 = wmma_f16(pa, bv, ao0); am0 = wmma_f16(pa, bm, am0);
        ivd = binv[cv + 1 * 16 + l15];
        bfrag_v(qk, qkm, ivd, cv, jb, 1, hf, l15, bv, bm);
        ao1 = wmma_f16(pa, bv, ao1); am1 = wmma_f16(pa, bm, am1);
        ivd = binv[cv + 2 * 16 + l15];
        bfrag_v(qk, qkm, ivd, cv, jb, 2, hf, l15, bv, bm);
        ao2 = wmma_f16(pa, bv, ao2); am2 = wmma_f16(pa, bm, am2);
        ivd = binv[cv + 3 * 16 + l15];
        bfrag_v(qk, qkm, ivd, cv, jb, 3, hf, l15, bv, bm);
        ao3 = wmma_f16(pa, bv, ao3); am3 = wmma_f16(pa, bm, am3);
    }

#pragma unroll
    for (int g = 0; g < 8; ++g) {
        int row = m0 + g + hf * 8;
        if (row < SEQ) {
            float u  = upd[b * SEQ + row];
            float is = u / rs[g];
            size_t base = ((size_t)b * SEQ + row) * DIM + h * DHEAD + l15;
            out_attn[base]      = ao0[g] * is;
            out_attn[base + 16] = ao1[g] * is;
            out_attn[base + 32] = ao2[g] * is;
            out_attn[base + 48] = ao3[g] * is;
            m_attn[base]      = am0[g] * is;
            m_attn[base + 16] = am1[g] * is;
            m_attn[base + 32] = am2[g] * is;
            m_attn[base + 48] = am3[g] * is;
        }
    }
}

// ---------------------------------------------------------------------------
// Overlap blending: ob/mb = oa/ma plus 2x2-avg cross additions.
// ---------------------------------------------------------------------------
__global__ __launch_bounds__(256) void blend_kernel(
    const float* __restrict__ oa, const float* __restrict__ ma,
    const float* __restrict__ upd, float* __restrict__ ob, float* __restrict__ mb)
{
    size_t idx = (size_t)blockIdx.x * 256 + threadIdx.x;
    if (idx >= (size_t)B_ * SEQ * DIM) return;
    int c = (int)(idx % DIM);
    size_t bt = idx / DIM;
    int t  = (int)(bt % SEQ);
    int bb = (int)(bt / SEQ);
    float vo = oa[idx], vm = ma[idx];
    if (t >= 1 && t <= 256) {
        // origin: avg2x2(pad(tail 15x15)) added to 16x16 grid region
        int r = t - 1, i = r >> 4, j = r & 15;
        float so = 0.0f, sm = 0.0f;
        for (int a = i; a <= i + 1; ++a)
            for (int b2 = j; b2 <= j + 1; ++b2)
                if (a >= 1 && a <= 15 && b2 >= 1 && b2 <= 15) {
                    size_t q = ((size_t)bb * SEQ + 257 + (a - 1) * 15 + (b2 - 1)) * DIM + c;
                    so += oa[q]; sm += ma[q];
                }
        float f = 0.25f * (1.0f - upd[bb * SEQ + t]);
        vo += so * f; vm += sm * f;
    } else if (t >= 257) {
        // bridge: avg2x2(grid 16x16) added to 15x15 tail region
        int q = t - 257, i = q / 15, j = q % 15;
        float so = 0.0f, sm = 0.0f;
        for (int a = i; a <= i + 1; ++a)
            for (int b2 = j; b2 <= j + 1; ++b2) {
                size_t q2 = ((size_t)bb * SEQ + 1 + a * 16 + b2) * DIM + c;
                so += oa[q2]; sm += ma[q2];
            }
        float f = 0.25f * (1.0f - upd[bb * SEQ + t]);
        vo += so * f; vm += sm * f;
    }
    ob[idx] = vo; mb[idx] = vm;
}

// mm[b*SEQ+t] = mean_c mb[b][t][c]
__global__ __launch_bounds__(256) void mm_kernel(
    const float* __restrict__ mb, float* __restrict__ mm)
{
    int t = blockIdx.x;
    __shared__ float red[8];
    const float* p = mb + (size_t)t * DIM;
    float s = 0.0f;
    for (int c = threadIdx.x; c < DIM; c += 256) s += p[c];
#pragma unroll
    for (int m = 1; m < 32; m <<= 1) s += __shfl_xor(s, m, 32);
    if ((threadIdx.x & 31) == 0) red[threadIdx.x >> 5] = s;
    __syncthreads();
    if (threadIdx.x == 0) {
        float tot = 0.0f;
        for (int i = 0; i < 8; ++i) tot += red[i];
        mm[t] = tot * (1.0f / (float)DIM);
    }
}

// inter[b][y][x]: NN upsample of grid means, blended with shifted means.
__global__ __launch_bounds__(256) void inter_kernel(
    const float* __restrict__ mm, float* __restrict__ inter)
{
    int idx = blockIdx.x * 256 + threadIdx.x;
    if (idx >= B_ * 256 * 256) return;
    int x = idx & 255, y = (idx >> 8) & 255, b = idx >> 16;
    float g = mm[b * SEQ + 1 + (y >> 4) * 16 + (x >> 4)];
    if (y >= 8 && y < 248 && x >= 8 && x < 248) {
        float s = mm[b * SEQ + 257 + ((y - 8) >> 4) * 15 + ((x - 8) >> 4)];
        g = 0.5f * (g + s);
    }
    inter[idx] = g;
}

// ---------------------------------------------------------------------------
extern "C" void kernel_launch(void* const* d_in, const int* in_sizes, int n_in,
                              void* d_out, int out_size, void* d_ws, size_t ws_size,
                              hipStream_t stream)
{
    (void)in_sizes; (void)n_in; (void)out_size; (void)ws_size;
    const float* x    = (const float*)d_in[0];
    const float* mask = (const float*)d_in[1];
    const float* Wqkv = (const float*)d_in[2];
    const float* Wout = (const float*)d_in[3];
    const float* bout = (const float*)d_in[4];

    const size_t nTok = (size_t)B_ * SEQ;   // 30848 (multiple of 32)

    float* ws   = (float*)d_ws;
    float* qkv  = ws;                        // nTok*1536
    float* qkvm = qkv  + nTok * QKVC;        // nTok*1536
    float* inv  = qkvm + nTok * QKVC;        // 64*1536
    float* upd  = inv  + (size_t)B_ * QKVC;  // nTok
    float* oa   = upd  + nTok;               // nTok*512
    float* ma   = oa   + nTok * DIM;         // nTok*512
    float* ob   = ma   + nTok * DIM;         // nTok*512
    float* mb   = ob   + nTok * DIM;         // nTok*512
    float* mmv  = mb   + nTok * DIM;         // nTok

    float* outF  = (float*)d_out;
    float* mF    = outF + nTok * DIM;
    float* inter = mF   + nTok * DIM;

    // 1) qkv = x @ Wqkv^T ;  qkv_m = mask @ |Wqkv|^T   (32x64 tiles per wave)
    {
        long tiles  = (long)(nTok >> 5) * (QKVC >> 6);
        int  blocks = (int)((tiles + 7) / 8);
        gemm_wmma<<<blocks, 256, 0, stream>>>(x,    Wqkv, nullptr, qkv,  (int)nTok, QKVC, DIM, 0);
        gemm_wmma<<<blocks, 256, 0, stream>>>(mask, Wqkv, nullptr, qkvm, (int)nTok, QKVC, DIM, 1);
    }
    // 2) per-(b,col) max -> reciprocal normalizers
    colinv_kernel<<<(B_ * QKVC + 255) / 256, 256, 0, stream>>>(qkvm, inv);
    // 3) updated flags
    updated_kernel<<<(int)nTok, 256, 0, stream>>>(mask, upd);
    // 4) attention (flash-style, WMMA)
    attn_kernel<<<B_ * HEADS * 4, 256, 0, stream>>>(qkv, qkvm, inv, upd, oa, ma);
    // 5) overlap blending
    blend_kernel<<<(int)((nTok * DIM + 255) / 256), 256, 0, stream>>>(oa, ma, upd, ob, mb);
    // 6) row means of blended m
    mm_kernel<<<(int)nTok, 256, 0, stream>>>(mb, mmv);
    // 7) inter upsample
    inter_kernel<<<(B_ * 256 * 256) / 256, 256, 0, stream>>>(mmv, inter);
    // 8) out_final = ob @ Wout^T + bout ;  m_final = mb @ |Wout|^T
    {
        long tiles  = (long)(nTok >> 5) * (DIM >> 6);
        int  blocks = (int)((tiles + 7) / 8);
        gemm_wmma<<<blocks, 256, 0, stream>>>(ob, Wout, bout,    outF, (int)nTok, DIM, DIM, 0);
        gemm_wmma<<<blocks, 256, 0, stream>>>(mb, Wout, nullptr, mF,   (int)nTok, DIM, DIM, 1);
    }
}
